// NonLocalBlock_18640158065120
// MI455X (gfx1250) — compile-verified, matrix operand order checked by reading, not run on hardware
//
#include <hip/hip_runtime.h>
#include <cstdint>

// Problem constants (fixed by the reference)
#define BATCH 8
#define CHN   512
#define NPIX  4096     // 64*64
#define NGRP  32
#define CPG   16       // channels per group
#define ATTN_SCALE 0.04419417382415922f   // 512^-0.5
#define LOG2E 1.4426950408889634f

typedef __attribute__((ext_vector_type(16))) __bf16 v16bf;
typedef __attribute__((ext_vector_type(8)))  float  v8f;

union FragAB { v16bf v; uint32_t u[8]; };

__device__ __forceinline__ uint16_t f2bf(float f) {
    uint32_t u = __float_as_uint(f);
    uint32_t r = u + 0x7FFFu + ((u >> 16) & 1u);   // round-to-nearest-even
    return (uint16_t)(r >> 16);
}
__device__ __forceinline__ uint32_t pack2bf(float lo, float hi) {
    return (uint32_t)f2bf(lo) | ((uint32_t)f2bf(hi) << 16);
}
// 16-bit A/B fragment layout: VGPR-pair t (t=0..7), half-wave hi -> K base of the
// packed bf16 pair (K, K+1). Matches CDNA5 ISA 16-bit A 16x32 table.
__device__ __forceinline__ int kpair(int t, int hi) {
    return ((t < 4) ? (2 * t) : (16 + 2 * (t - 4))) + 8 * hi;
}

// Load one A/B fragment whose packed pairs are contiguous in memory at
// base + 16*hi bytes and base + 32 + 16*hi bytes (merges to 2x b128 loads).
__device__ __forceinline__ FragAB load_frag_row(const uint16_t* rowbase, int hi) {
    FragAB f;
#pragma unroll
    for (int t = 0; t < 8; ++t)
        f.u[t] = *(const uint32_t*)(rowbase + kpair(t, hi));
    return f;
}

// ---------------------------------------------------------------------------
// Kernel 1: GroupNorm, write normalized activations token-major bf16 Ht[b,n,c]
// grid = B*NGRP blocks, 256 threads
// ---------------------------------------------------------------------------
__global__ __launch_bounds__(256) void gn_kernel(const float* __restrict__ x,
                                                 const float* __restrict__ gsc,
                                                 const float* __restrict__ gbs,
                                                 uint16_t* __restrict__ Ht) {
    const int b = blockIdx.x >> 5;
    const int g = blockIdx.x & 31;
    const int tid = threadIdx.x;
    const float* xp = x + ((size_t)(b * CHN + g * CPG)) * NPIX;

    float s = 0.f, s2 = 0.f;
    for (int idx = tid; idx < CPG * NPIX; idx += 256) {
        float v = xp[idx];
        s += v; s2 += v * v;
    }
    __shared__ float red[512];
    red[tid] = s; red[256 + tid] = s2;
    __syncthreads();
    for (int off = 128; off > 0; off >>= 1) {
        if (tid < off) {
            red[tid] += red[tid + off];
            red[256 + tid] += red[256 + tid + off];
        }
        __syncthreads();
    }
    const float inv_n = 1.0f / (CPG * NPIX);
    float mean = red[0] * inv_n;
    float var  = red[256] * inv_n - mean * mean;
    float rinv = rsqrtf(var + 1e-6f);

    for (int idx = tid; idx < CPG * NPIX; idx += 256) {
        int cl = idx >> 12;            // idx / NPIX
        int n  = idx & (NPIX - 1);
        int c  = g * CPG + cl;
        float v = (xp[idx] - mean) * rinv * gsc[c] + gbs[c];
        Ht[((size_t)b * NPIX + n) * CHN + c] = f2bf(v);
    }
}

// ---------------------------------------------------------------------------
// Kernel 2: QKV projection.  Qt/Kt/Vt[b,n,o] = Ht[b,n,:] @ W^T + bias (bf16)
// Q pre-scaled by C^-0.5.  grid = B * (NPIX/128), 256 threads (8 waves).
// A-fragments streamed from global (L2-resident); B-fragments double-buffered
// from LDS so each WMMA waits on a load issued one full WMMA earlier.
// ---------------------------------------------------------------------------
__global__ __launch_bounds__(256) void qkv_kernel(const uint16_t* __restrict__ Ht,
                                                  const float* __restrict__ wq, const float* __restrict__ bq,
                                                  const float* __restrict__ wk, const float* __restrict__ bk,
                                                  const float* __restrict__ wv, const float* __restrict__ bv,
                                                  uint16_t* __restrict__ Qt,
                                                  uint16_t* __restrict__ Kt,
                                                  uint16_t* __restrict__ Vt) {
    const int b  = blockIdx.x >> 5;
    const int n0 = (blockIdx.x & 31) * 128;
    const int tid  = threadIdx.x;
    const int wave = tid >> 5;
    const int lane = tid & 31;
    const int l16  = lane & 15;
    const int hi   = lane >> 4;

    __shared__ uint32_t slab[CHN * 16];   // 32KB: [o][16 bf16-pairs] of one 32-wide K slab

    const uint16_t* hrow = Ht + ((size_t)b * NPIX + n0 + wave * 16 + l16) * CHN;

    const float* wptr[3] = {wq, wk, wv};
    const float* bptr[3] = {bq, bk, bv};
    uint16_t*    optr[3] = {Qt, Kt, Vt};

    auto loadB = [&](int ot) {
        FragAB bf;
#pragma unroll
        for (int t = 0; t < 8; ++t)
            bf.u[t] = slab[(ot * 16 + l16) * 16 + (kpair(t, hi) >> 1)];
        return bf;
    };

    for (int w = 0; w < 3; ++w) {
        v8f acc[32];
#pragma unroll
        for (int ot = 0; ot < 32; ++ot) {
            float bb = bptr[w][ot * 16 + l16];
#pragma unroll
            for (int r = 0; r < 8; ++r) acc[ot][r] = bb;
        }
        for (int c0 = 0; c0 < 16; ++c0) {
            __syncthreads();
            // stage W[:, c0*32 : c0*32+32] as bf16 pairs, layout slab[o*16 + pair]
            const float* wbase = wptr[w] + c0 * 32;
#pragma unroll
            for (int k = 0; k < 32; ++k) {
                int p = tid + k * 256;            // p = o*16 + t
                int o = p >> 4, t = p & 15;
                const float* ws = wbase + (size_t)o * CHN + 2 * t;
                slab[p] = pack2bf(ws[0], ws[1]);
            }
            __syncthreads();
            // stream this chunk's A fragment from global (L2 hit)
            FragAB af = load_frag_row(hrow + c0 * 32, hi);
            FragAB bfb[2];
            bfb[0] = loadB(0);
#pragma unroll
            for (int ot = 0; ot < 32; ++ot) {
                if (ot + 1 < 32) bfb[(ot + 1) & 1] = loadB(ot + 1);
                acc[ot] = __builtin_amdgcn_wmma_f32_16x16x32_bf16(
                    false, af.v, false, bfb[ot & 1].v, (short)0, acc[ot], false, false);
            }
        }
        const float mult = (w == 0) ? ATTN_SCALE : 1.0f;
        uint16_t* op = optr[w];
#pragma unroll
        for (int ot = 0; ot < 32; ++ot) {
            int o = ot * 16 + l16;
#pragma unroll
            for (int r = 0; r < 8; ++r) {
                int row = n0 + wave * 16 + r + 8 * hi;
                op[((size_t)b * NPIX + row) * CHN + o] = f2bf(acc[ot][r] * mult);
            }
        }
    }
}

// ---------------------------------------------------------------------------
// Kernel 3: flash attention + residual.
// grid = B * (NPIX/128), 256 threads (8 waves). Wave owns 16 query rows and a
// full 16x512 f32 O accumulator (256 VGPRs). Key blocks of 32 staged through
// one 32KB LDS buffer (K phase: [j][c]; V phase: transposed [c][j]).
// Q fragments prefetched 3 chunks ahead (hides L2 latency); K/V B-fragments
// double-buffered from LDS.
// ---------------------------------------------------------------------------
__global__ __launch_bounds__(256) void attn_kernel(const uint16_t* __restrict__ Qt,
                                                   const uint16_t* __restrict__ Kt,
                                                   const uint16_t* __restrict__ Vt,
                                                   const float* __restrict__ x,
                                                   float* __restrict__ out) {
    const int b  = blockIdx.x >> 5;
    const int i0 = (blockIdx.x & 31) * 128;
    const int tid  = threadIdx.x;
    const int wave = tid >> 5;
    const int lane = tid & 31;
    const int l16  = lane & 15;
    const int hi   = lane >> 4;

    __shared__ uint32_t kvbuf[32 * 256];   // 32KB: K [32 j][256 pairs] / V [512 c][16 pairs]
    __shared__ uint32_t ptile[8 * 256];    // 8KB : per-wave P tile [16 i][16 j-pairs]
    uint16_t* kv16 = (uint16_t*)kvbuf;
    uint16_t* pt16 = (uint16_t*)ptile;

    const uint16_t* qrow = Qt + ((size_t)b * NPIX + i0 + wave * 16 + l16) * CHN;

    auto loadBK = [&](int c0, int jt) {      // K-phase B fragment (K=c, N=j)
        FragAB bf;
#pragma unroll
        for (int t = 0; t < 8; ++t)
            bf.u[t] = kvbuf[(jt * 16 + l16) * 256 + c0 * 16 + (kpair(t, hi) >> 1)];
        return bf;
    };
    auto loadBV = [&](int ct) {              // V-phase B fragment (K=j, N=c)
        FragAB bf;
#pragma unroll
        for (int t = 0; t < 8; ++t)
            bf.u[t] = kvbuf[(ct * 16 + l16) * 16 + (kpair(t, hi) >> 1)];
        return bf;
    };

    v8f oacc[32];
#pragma unroll
    for (int ct = 0; ct < 32; ++ct)
#pragma unroll
        for (int r = 0; r < 8; ++r) oacc[ct][r] = 0.f;

    float mrun[8], lrun[8];
#pragma unroll
    for (int r = 0; r < 8; ++r) { mrun[r] = -3.4e38f; lrun[r] = 0.f; }

    for (int jb = 0; jb < NPIX / 32; ++jb) {
        const int j0 = jb * 32;
        __syncthreads();                                   // kvbuf reuse fence
        // ---- stage K block [32 j][512 c] (row-contiguous b128 copies) ----
        {
            const uint4* src = (const uint4*)(Kt + ((size_t)b * NPIX + j0) * CHN);
            uint4* dst = (uint4*)kvbuf;
#pragma unroll
            for (int k = 0; k < 8; ++k) dst[tid + k * 256] = src[tid + k * 256];
        }
        __syncthreads();

        // ---- S = Q K^T tile (16 x 32); Q frags prefetched 3 chunks ahead ----
        v8f sacc[2];
#pragma unroll
        for (int jt = 0; jt < 2; ++jt)
#pragma unroll
            for (int r = 0; r < 8; ++r) sacc[jt][r] = 0.f;

        FragAB qfb[4];
#pragma unroll
        for (int p = 0; p < 3; ++p) qfb[p] = load_frag_row(qrow + p * 32, hi);
#pragma unroll
        for (int c0 = 0; c0 < 16; ++c0) {
            if (c0 + 3 < 16) qfb[(c0 + 3) & 3] = load_frag_row(qrow + (c0 + 3) * 32, hi);
            FragAB bf0 = loadBK(c0, 0);
            FragAB bf1 = loadBK(c0, 1);
            sacc[0] = __builtin_amdgcn_wmma_f32_16x16x32_bf16(
                false, qfb[c0 & 3].v, false, bf0.v, (short)0, sacc[0], false, false);
            sacc[1] = __builtin_amdgcn_wmma_f32_16x16x32_bf16(
                false, qfb[c0 & 3].v, false, bf1.v, (short)0, sacc[1], false, false);
        }

        // ---- online softmax (rows live in VGPR r, cols across 16 lanes) ----
#pragma unroll
        for (int r = 0; r < 8; ++r) {
            float mx = fmaxf(sacc[0][r], sacc[1][r]);
#pragma unroll
            for (int m = 1; m < 16; m <<= 1) mx = fmaxf(mx, __shfl_xor(mx, m, 32));
            float mnew  = fmaxf(mrun[r], mx);
            float alpha = exp2f((mrun[r] - mnew) * LOG2E);
            float ps = 0.f;
#pragma unroll
            for (int jt = 0; jt < 2; ++jt) {
                float p = exp2f((sacc[jt][r] - mnew) * LOG2E);
                sacc[jt][r] = p;
                ps += p;
            }
#pragma unroll
            for (int m = 1; m < 16; m <<= 1) ps += __shfl_xor(ps, m, 32);
            lrun[r] = lrun[r] * alpha + ps;
            mrun[r] = mnew;
#pragma unroll
            for (int ct = 0; ct < 32; ++ct) oacc[ct][r] *= alpha;
        }

        // ---- P (D layout) -> LDS as bf16 ----
#pragma unroll
        for (int jt = 0; jt < 2; ++jt)
#pragma unroll
            for (int r = 0; r < 8; ++r)
                pt16[wave * 512 + (r + 8 * hi) * 32 + jt * 16 + l16] = f2bf(sacc[jt][r]);

        __syncthreads();                                   // P visible; kvbuf free
        // ---- stage V block transposed: kv16[c*32 + j] = Vt[j][c] ----
        {
            const uint32_t* src = (const uint32_t*)(Vt + ((size_t)b * NPIX + j0) * CHN);
#pragma unroll
            for (int k = 0; k < 32; ++k) {
                int idx = tid + k * 256;     // 0..8191
                int j = idx >> 8, cp = idx & 255;
                uint32_t v = src[(size_t)j * 256 + cp];
                kv16[(2 * cp) * 32 + j]     = (uint16_t)(v & 0xFFFF);
                kv16[(2 * cp + 1) * 32 + j] = (uint16_t)(v >> 16);
            }
        }
        __syncthreads();

        // ---- O += P @ V^T (B fragments double-buffered) ----
        FragAB pf;
#pragma unroll
        for (int t = 0; t < 8; ++t)
            pf.u[t] = ptile[wave * 256 + l16 * 16 + (kpair(t, hi) >> 1)];
        FragAB bvb[2];
        bvb[0] = loadBV(0);
#pragma unroll
        for (int ct = 0; ct < 32; ++ct) {
            if (ct + 1 < 32) bvb[(ct + 1) & 1] = loadBV(ct + 1);
            oacc[ct] = __builtin_amdgcn_wmma_f32_16x16x32_bf16(
                false, pf.v, false, bvb[ct & 1].v, (short)0, oacc[ct], false, false);
        }
    }

    // ---- finalize: O/l + residual, out[b,c,i] ----
    float inv[8];
#pragma unroll
    for (int r = 0; r < 8; ++r) inv[r] = 1.0f / lrun[r];
#pragma unroll
    for (int ct = 0; ct < 32; ++ct) {
        int c = ct * 16 + l16;
#pragma unroll
        for (int r = 0; r < 8; ++r) {
            int row = i0 + wave * 16 + r + 8 * hi;
            size_t idx = ((size_t)b * CHN + c) * NPIX + row;
            out[idx] = x[idx] + oacc[ct][r] * inv[r];
        }
    }
}

// ---------------------------------------------------------------------------
extern "C" void kernel_launch(void* const* d_in, const int* in_sizes, int n_in,
                              void* d_out, int out_size, void* d_ws, size_t ws_size,
                              hipStream_t stream) {
    const float* x   = (const float*)d_in[0];
    const float* gsc = (const float*)d_in[1];
    const float* gbs = (const float*)d_in[2];
    const float* wq  = (const float*)d_in[3];
    const float* bq  = (const float*)d_in[4];
    const float* wk  = (const float*)d_in[5];
    const float* bk  = (const float*)d_in[6];
    const float* wv  = (const float*)d_in[7];
    const float* bv  = (const float*)d_in[8];

    const size_t elems = (size_t)BATCH * NPIX * CHN;   // 16,777,216 bf16 each
    uint16_t* Ht = (uint16_t*)d_ws;
    uint16_t* Qt = Ht + elems;
    uint16_t* Kt = Qt + elems;
    uint16_t* Vt = Kt + elems;   // total workspace: 128 MB

    gn_kernel<<<BATCH * NGRP, 256, 0, stream>>>(x, gsc, gbs, Ht);
    qkv_kernel<<<BATCH * (NPIX / 128), 256, 0, stream>>>(Ht, wq, bq, wk, bk, wv, bv, Qt, Kt, Vt);
    attn_kernel<<<BATCH * (NPIX / 128), 256, 0, stream>>>(Qt, Kt, Vt, x, (float*)d_out);
}
